// Decoder_45122926412481
// MI455X (gfx1250) — compile-verified, matrix operand order checked by reading, not run on hardware
//
#include <hip/hip_runtime.h>
#include <hip/hip_bf16.h>
#include <math.h>

// ---------------- problem dims ----------------
#define BB   32
#define LL   64
#define HH   512
#define WD   300
#define VV   32000
#define TT   48
#define KIH  812          // H + WD
#define KIHP 832          // padded to 32
#define KOUT 1324         // 2H + WD
#define KOUTP 1344        // padded to 32 (42 chunks)
#define G3H  1536         // 3H

typedef __attribute__((ext_vector_type(16))) _Float16 v16h;
typedef __attribute__((ext_vector_type(8)))  _Float16 v8h;
typedef __attribute__((ext_vector_type(8)))  float    v8f;

// ---------------- WMMA helpers ----------------
// 16-bit A (16x32) / B^T tile lane layout (ISA 7.12.2):
// lanes 0-15 hold K {0..7, 16..23}; lanes 16-31 hold K {8..15, 24..31}
__device__ __forceinline__ int wmma_kbase(int lane) { return (lane >> 4) << 3; }

__device__ __forceinline__ v16h load_row_tile(const _Float16* rowPtr, int kc, int lane) {
  const _Float16* p = rowPtr + kc + wmma_kbase(lane);
  v8h lo = *(const v8h*)(p);
  v8h hi = *(const v8h*)(p + 16);
  v16h r;
#pragma unroll
  for (int j = 0; j < 8; ++j) { r[j] = lo[j]; r[j + 8] = hi[j]; }
  return r;
}

__device__ __forceinline__ v8f wmma_f16(v16h a, v16h b, v8f c) {
  return __builtin_amdgcn_wmma_f32_16x16x32_f16(false, a, false, b, (short)0, c, false, false);
}

// C/D f32 16x16 layout: lane n = lane&15 ; VGPR r -> row r + (lane>=16 ? 8 : 0)
__device__ __forceinline__ void store_ctile(float* C, int ldc, int m0, int n0, int lane,
                                            v8f acc, const float* bias) {
  int n = n0 + (lane & 15);
  int mb = m0 + ((lane >> 4) << 3);
  float bv = bias ? bias[n] : 0.f;
#pragma unroll
  for (int r = 0; r < 8; ++r) C[(size_t)(mb + r) * ldc + n] = acc[r] + bv;
}

// ---------------- generic WMMA GEMM ----------------
// C[m,n] = sum_k A[m,k] * W[n,k] + bias[n]
// (W stored "N-major": row n holds K contiguously -> both A and W tiles are
//  two global_load_b128 per lane per 16x16x32 chunk)
__global__ __launch_bounds__(256) void k_gemm_bt(const _Float16* __restrict__ A, int lda,
                                                 const _Float16* __restrict__ W, int ldw,
                                                 const float* __restrict__ bias,
                                                 float* __restrict__ C, int ldc,
                                                 int Mtiles, int Ntiles, int Kchunks) {
  int lane = threadIdx.x & 31;
  int wave = (blockIdx.x * blockDim.x + threadIdx.x) >> 5;
  if (wave >= Mtiles * Ntiles) return;
  int mt = wave % Mtiles, nt = wave / Mtiles;
  int m0 = mt * 16, n0 = nt * 16;
  const _Float16* arow = A + (size_t)(m0 + (lane & 15)) * lda;
  const _Float16* wrow = W + (size_t)(n0 + (lane & 15)) * ldw;
  v8f acc = {};
  for (int kc = 0; kc < Kchunks * 32; kc += 32) {
    v16h a = load_row_tile(arow, kc, lane);
    v16h b = load_row_tile(wrow, kc, lane);
    acc = wmma_f16(a, b, acc);
  }
  store_ctile(C, ldc, m0, n0, lane, acc, bias);
}

// Output projection: logits(32 x 32000) = Y(32 x 1344) * Wout^T + b_out.
// Each wave owns one 16-wide vocab tile and BOTH batch tiles -> Wout read once.
__global__ __launch_bounds__(256) void k_out_gemm(const _Float16* __restrict__ Y,
                                                  const _Float16* __restrict__ Wout,
                                                  const float* __restrict__ bias,
                                                  float* __restrict__ logits) {
  int lane = threadIdx.x & 31;
  int wave = (blockIdx.x * blockDim.x + threadIdx.x) >> 5;
  if (wave >= VV / 16) return;
  int n0 = wave * 16;
  const _Float16* wrow = Wout + (size_t)(n0 + (lane & 15)) * KOUTP;
  const _Float16* a0 = Y + (size_t)(lane & 15) * KOUTP;
  const _Float16* a1 = Y + (size_t)(16 + (lane & 15)) * KOUTP;
  v8f acc0 = {}, acc1 = {};
  for (int kc = 0; kc < KOUTP; kc += 32) {
    if (kc + 288 <= KOUTP) __builtin_prefetch(wrow + kc + 256, 0, 1);
    v16h b  = load_row_tile(wrow, kc, lane);
    v16h x0 = load_row_tile(a0, kc, lane);
    v16h x1 = load_row_tile(a1, kc, lane);
    acc0 = wmma_f16(x0, b, acc0);
    acc1 = wmma_f16(x1, b, acc1);
  }
  store_ctile(logits, VV, 0,  n0, lane, acc0, bias);
  store_ctile(logits, VV, 16, n0, lane, acc1, bias);
}

// ---------------- support kernels ----------------
__global__ void k_cvt(const float* __restrict__ src, _Float16* __restrict__ dst,
                      int cols, int dcols) {
  int c = blockIdx.x * blockDim.x + threadIdx.x;
  int r = blockIdx.y;
  if (c < dcols)
    dst[(size_t)r * dcols + c] = (c < cols) ? (_Float16)src[(size_t)r * cols + c] : (_Float16)0.f;
}

// transpose + f16: dst[n, k] = src[k, n]; src is rows x cols row-major,
// dst is cols x rows row-major. Used so ALL GEMMs take the contiguous path.
__global__ void k_cvt_t(const float* __restrict__ src, _Float16* __restrict__ dst,
                        int rows, int cols) {
  int k = blockIdx.x * blockDim.x + threadIdx.x;   // source row
  int n = blockIdx.y;                              // source col
  if (k < rows)
    dst[(size_t)n * rows + k] = (_Float16)src[(size_t)k * cols + n];
}

__global__ void k_emb(const int* __restrict__ targets, const float* __restrict__ embed,
                      float* __restrict__ emb) {
  int bt = blockIdx.x;                 // b*TT + t
  int b = bt / TT, t = bt % TT;
  int tok = (t == 0) ? 1 : targets[b * TT + t - 1];  // SOS_ID = 1
  for (int j = threadIdx.x; j < WD; j += blockDim.x)
    emb[(size_t)bt * WD + j] = embed[(size_t)tok * WD + j];
}

__global__ void k_copy_f32(const float* __restrict__ src, float* __restrict__ dst, size_t n) {
  for (size_t i = blockIdx.x * (size_t)blockDim.x + threadIdx.x; i < n;
       i += (size_t)gridDim.x * blockDim.x) dst[i] = src[i];
}

__global__ void k_zero_f32(float* __restrict__ dst, size_t n) {
  for (size_t i = blockIdx.x * (size_t)blockDim.x + threadIdx.x; i < n;
       i += (size_t)gridDim.x * blockDim.x) dst[i] = 0.f;
}

// attention energies + softmax + context; emits x16 = f16([ctx, emb_t, 0pad])
__global__ __launch_bounds__(256) void k_attn(const float* __restrict__ encproj,
                                              const float* __restrict__ hq,
                                              const float* __restrict__ venergy,
                                              const float* __restrict__ enc,
                                              const float* __restrict__ emb, int t,
                                              _Float16* __restrict__ x16) {
  int b = blockIdx.x, tid = threadIdx.x;
  __shared__ float part[LL][4];
  __shared__ float sc[LL];
  __shared__ float red0;
  int l = tid & 63, seg = tid >> 6;                       // 64 l's x 4 H-segments
  const float* ep  = encproj + ((size_t)b * LL + l) * HH + seg * 128;
  const float* hqb = hq + b * HH + seg * 128;
  const float* vb  = venergy + seg * 128;
  float s = 0.f;
  for (int h = 0; h < 128; ++h) s += vb[h] * tanhf(ep[h] + hqb[h]);
  part[l][seg] = s;
  __syncthreads();
  if (tid < LL) sc[tid] = part[tid][0] + part[tid][1] + part[tid][2] + part[tid][3];
  __syncthreads();
  if (tid == 0) {
    float m = sc[0];
    for (int i = 1; i < LL; ++i) m = fmaxf(m, sc[i]);
    float ssum = 0.f;
    for (int i = 0; i < LL; ++i) { sc[i] = expf(sc[i] - m); ssum += sc[i]; }
    red0 = ssum;
  }
  __syncthreads();
  float inv = 1.f / red0;
  for (int h = tid; h < HH; h += blockDim.x) {
    float c = 0.f;
    for (int l2 = 0; l2 < LL; ++l2) c += sc[l2] * enc[((size_t)b * LL + l2) * HH + h];
    x16[(size_t)b * KIHP + h] = (_Float16)(c * inv);
  }
  for (int j = tid; j < WD; j += blockDim.x)
    x16[(size_t)b * KIHP + HH + j] = (_Float16)emb[((size_t)b * TT + t) * WD + j];
  for (int j = tid; j < KIHP - KIH; j += blockDim.x)
    x16[(size_t)b * KIHP + KIH + j] = (_Float16)0.f;
}

// GRU gate math + emits y16 = f16([h_new, emb_t, hid_old, 0pad])
__global__ void k_gru(const float* __restrict__ gx, const float* __restrict__ gh,
                      const float* __restrict__ hid_old, const float* __restrict__ emb,
                      int t, float* __restrict__ hid_new, _Float16* __restrict__ y16) {
  int i = blockIdx.x * blockDim.x + threadIdx.x;
  if (i >= BB * HH) return;
  int b = i >> 9, j = i & (HH - 1);
  const float* gxb = gx + (size_t)b * G3H;
  const float* ghb = gh + (size_t)b * G3H;
  float r = 1.f / (1.f + expf(-(gxb[j] + ghb[j])));
  float z = 1.f / (1.f + expf(-(gxb[HH + j] + ghb[HH + j])));
  float n = tanhf(gxb[2 * HH + j] + r * ghb[2 * HH + j]);
  float ho = hid_old[i];
  float hn = (1.f - z) * n + z * ho;
  hid_new[i] = hn;
  _Float16* yb = y16 + (size_t)b * KOUTP;
  yb[j]            = (_Float16)hn;
  yb[KIH + j]      = (_Float16)ho;          // hid_old at cols 812..1323
  if (j < WD) yb[HH + j] = (_Float16)emb[((size_t)b * TT + t) * WD + j];
  if (j < KOUTP - KOUT) yb[KOUT + j] = (_Float16)0.f;
}

__global__ __launch_bounds__(256) void k_logsoftmax(const float* __restrict__ logits,
                                                    float* __restrict__ out, int t) {
  int b = blockIdx.x, tid = threadIdx.x;
  __shared__ float red[256];
  const float* row = logits + (size_t)b * VV;
  float m = -INFINITY;
  for (int v = tid; v < VV; v += 256) m = fmaxf(m, row[v]);
  red[tid] = m; __syncthreads();
  for (int s = 128; s > 0; s >>= 1) { if (tid < s) red[tid] = fmaxf(red[tid], red[tid + s]); __syncthreads(); }
  m = red[0]; __syncthreads();
  float sum = 0.f;
  for (int v = tid; v < VV; v += 256) sum += expf(row[v] - m);
  red[tid] = sum; __syncthreads();
  for (int s = 128; s > 0; s >>= 1) { if (tid < s) red[tid] += red[tid + s]; __syncthreads(); }
  float lse = m + logf(red[0]);
  float* orow = out + ((size_t)t * BB + b) * VV;
  for (int v = tid; v < VV; v += 256) orow[v] = row[v] - lse;
}

// ---------------- host orchestration ----------------
extern "C" void kernel_launch(void* const* d_in, const int* in_sizes, int n_in,
                              void* d_out, int out_size, void* d_ws, size_t ws_size,
                              hipStream_t stream) {
  (void)in_sizes; (void)n_in; (void)out_size; (void)ws_size;
  const float* enc      = (const float*)d_in[0];   // B,L,H
  const float* ctx0     = (const float*)d_in[1];   // 1,B,H
  const int*   targets  = (const int*)  d_in[2];   // B,T
  const float* embed    = (const float*)d_in[3];   // V,WD
  const float* W_attn   = (const float*)d_in[4];   // 2H,H
  const float* b_attn   = (const float*)d_in[5];   // H
  const float* v_energy = (const float*)d_in[6];   // H
  const float* W_ih     = (const float*)d_in[7];   // 3H,H+WD
  const float* W_hh     = (const float*)d_in[8];   // 3H,H
  const float* b_ih     = (const float*)d_in[9];
  const float* b_hh     = (const float*)d_in[10];
  const float* W_out    = (const float*)d_in[11];  // V,2H+WD
  const float* b_out    = (const float*)d_in[12];

  char* wsp = (char*)d_ws;
  auto alloc = [&](size_t bytes) -> void* {
    void* p = (void*)wsp;
    wsp += (bytes + 255) & ~(size_t)255;
    return p;
  };
  _Float16* wout16  = (_Float16*)alloc((size_t)VV * KOUTP * 2);   // 86.0 MB (L2-resident)
  _Float16* wih16   = (_Float16*)alloc((size_t)G3H * KIHP * 2);
  _Float16* whh16   = (_Float16*)alloc((size_t)G3H * HH * 2);
  _Float16* waencT  = (_Float16*)alloc((size_t)HH * HH * 2);      // W_a_enc^T (N-major)
  _Float16* wahidT  = (_Float16*)alloc((size_t)HH * HH * 2);      // W_a_hid^T (N-major)
  _Float16* enc16   = (_Float16*)alloc((size_t)BB * LL * HH * 2);
  float*    encproj = (float*)   alloc((size_t)BB * LL * HH * 4);
  float*    emb     = (float*)   alloc((size_t)BB * TT * WD * 4);
  float*    hidA    = (float*)   alloc((size_t)BB * HH * 4);
  float*    hidB    = (float*)   alloc((size_t)BB * HH * 4);
  _Float16* hid16   = (_Float16*)alloc((size_t)BB * HH * 2);
  float*    hq      = (float*)   alloc((size_t)BB * HH * 4);
  _Float16* x16     = (_Float16*)alloc((size_t)BB * KIHP * 2);
  _Float16* y16     = (_Float16*)alloc((size_t)BB * KOUTP * 2);
  float*    gx      = (float*)   alloc((size_t)BB * G3H * 4);
  float*    gh      = (float*)   alloc((size_t)BB * G3H * 4);
  float*    logits  = (float*)   alloc((size_t)BB * VV * 4);

  dim3 blk(256);
  // one-time: f16 weight copies / transposes (graph-captured; trivial vs 48x W_out stream)
  k_cvt<<<dim3((KOUTP + 255) / 256, VV),  blk, 0, stream>>>(W_out,  wout16,  KOUT, KOUTP);
  k_cvt<<<dim3((KIHP  + 255) / 256, G3H), blk, 0, stream>>>(W_ih,   wih16,   KIH,  KIHP);
  k_cvt<<<dim3(2, G3H),                   blk, 0, stream>>>(W_hh,   whh16,   HH, HH);
  k_cvt_t<<<dim3(2, HH),                  blk, 0, stream>>>(W_attn,                    waencT, HH, HH);
  k_cvt_t<<<dim3(2, HH),                  blk, 0, stream>>>(W_attn + (size_t)HH * HH,  wahidT, HH, HH);
  k_cvt<<<dim3(2, BB * LL),               blk, 0, stream>>>(enc,    enc16,   HH, HH);
  k_emb<<<dim3(BB * TT), blk, 0, stream>>>(targets, embed, emb);
  // enc_proj = enc(2048x512) @ W_a_enc(512x512)  (WMMA, 4096 waves)
  k_gemm_bt<<<dim3(512), blk, 0, stream>>>(enc16, HH, waencT, HH, nullptr,
                                           encproj, HH, (BB * LL) / 16, HH / 16, HH / 32);
  k_copy_f32<<<dim3(64), blk, 0, stream>>>(ctx0, hidA, (size_t)BB * HH);

  float* hcur = hidA;
  float* hnxt = hidB;
  for (int t = 0; t < TT; ++t) {
    k_cvt<<<dim3(2, BB), blk, 0, stream>>>(hcur, hid16, HH, HH);
    // hq = hid @ W_a_hid + b_attn
    k_gemm_bt<<<dim3(8), blk, 0, stream>>>(hid16, HH, wahidT, HH, b_attn, hq, HH,
                                           2, HH / 16, HH / 32);
    k_attn<<<dim3(BB), blk, 0, stream>>>(encproj, hq, v_energy, enc, emb, t, x16);
    // gx = x @ W_ih^T + b_ih ; gh = hid @ W_hh^T + b_hh
    k_gemm_bt<<<dim3(24), blk, 0, stream>>>(x16, KIHP, wih16, KIHP, b_ih, gx, G3H,
                                            2, G3H / 16, KIHP / 32);
    k_gemm_bt<<<dim3(24), blk, 0, stream>>>(hid16, HH, whh16, HH, b_hh, gh, G3H,
                                            2, G3H / 16, HH / 32);
    k_gru<<<dim3(64), blk, 0, stream>>>(gx, gh, hcur, emb, t, hnxt, y16);
    // big projection: 2000 vocab tiles, W_out f16 streamed from L2
    k_out_gemm<<<dim3(250), blk, 0, stream>>>(y16, wout16, b_out, logits);
    k_logsoftmax<<<dim3(BB), blk, 0, stream>>>(logits, (float*)d_out, t);
    float* tmp = hcur; hcur = hnxt; hnxt = tmp;
  }
  // zero tail rows t = 48..63 of (L,B,V) output, then append h_final
  k_zero_f32<<<dim3(4096), blk, 0, stream>>>((float*)d_out + (size_t)TT * BB * VV,
                                             (size_t)(LL - TT) * BB * VV);
  k_copy_f32<<<dim3(64), blk, 0, stream>>>(hcur, (float*)d_out + (size_t)LL * BB * VV,
                                           (size_t)BB * HH);
}
